// MultiHeadSelfAttention_37761352466487
// MI455X (gfx1250) — matrix operand order repaired and checked by reading, not compiled, not run on hardware
//
#include <hip/hip_runtime.h>

typedef __bf16 bf16_t;
typedef __attribute__((ext_vector_type(16))) __bf16 v16bf;
typedef __attribute__((ext_vector_type(8)))  float  v8f;

union Frag16 { v16bf v; uint4 u[2]; };

constexpr int D_MODEL   = 1024;
constexpr int NUM_HEADS = 16;
constexpr int HEAD_DIM  = 64;
constexpr int SEQ_LEN   = 2048;
constexpr int BATCH     = 2;
constexpr int MROWS     = BATCH * SEQ_LEN;   // 4096 GEMM rows

static __device__ __forceinline__ v8f wmma_bf16(v16bf a, v16bf b, v8f c) {
  // v_wmma_f32_16x16x32_bf16  D = A(16x32) * B(32x16) + C
  return __builtin_amdgcn_wmma_f32_16x16x32_bf16(false, a, false, b, (short)0, c,
                                                 false, false);
}

// Async copy of 32 contiguous bytes (per lane) from global to LDS.
// offset: immediate is added to BOTH the LDS and global addresses (ISA 10.7),
// so the second 16B chunk reuses the same address registers.
static __device__ __forceinline__ void async_copy_32B(unsigned lds_off,
                                                      const void* gptr) {
  unsigned long long ga = (unsigned long long)(size_t)gptr;
  asm volatile(
      "global_load_async_to_lds_b128 %0, %1, off\n\t"
      "global_load_async_to_lds_b128 %0, %1, off offset:16"
      :: "v"(lds_off), "v"(ga)
      : "memory");
}

static __device__ __forceinline__ void wait_async0() {
  asm volatile("s_wait_asynccnt 0x0" ::: "memory");
}

// ---------------------------------------------------------------- converts
__global__ void cvt_f32_to_bf16_kernel(const float* __restrict__ in,
                                       bf16_t* __restrict__ out, int n) {
  int i = blockIdx.x * blockDim.x + threadIdx.x;
  if (i < n) out[i] = (bf16_t)in[i];
}

// RoPE + bf16 convert for Q / K.  outScale folds the 1/sqrt(Dh) into Q.
__global__ void rope_qk_kernel(const float* __restrict__ in,
                               bf16_t* __restrict__ out,
                               const int* __restrict__ pos, float outScale) {
  const int PAIRS = D_MODEL / 2;                 // 512 pairs per token
  int idx = blockIdx.x * blockDim.x + threadIdx.x;
  if (idx >= BATCH * SEQ_LEN * PAIRS) return;
  int e2 = idx % PAIRS;
  int l  = (idx / PAIRS) % SEQ_LEN;
  int b  = idx / (PAIRS * SEQ_LEN);
  int h  = e2 >> 5;                               // 32 pairs per head
  int i  = e2 & 31;
  // inv_freq = 10000^(-(2i/64)) = exp(-(2i/64)*ln(10000))
  float invf = __expf(-((2.0f * (float)i) / 64.0f) * 9.210340371976184f);
  float ang  = (float)pos[l] * invf;
  float sn, cs;
  __sincosf(ang, &sn, &cs);
  size_t base = ((size_t)b * SEQ_LEN + l) * D_MODEL + h * HEAD_DIM + 2 * i;
  float a  = in[base];
  float bb = in[base + 1];
  out[base]     = (bf16_t)((a * cs - bb * sn) * outScale);
  out[base + 1] = (bf16_t)((a * sn + bb * cs) * outScale);
}

// V: fp32 (B,L,D) -> bf16 transposed (B,H,Dh,L) so attention B-fragments are
// contiguous along the key dimension.
__global__ void vtrans_kernel(const float* __restrict__ in,
                              bf16_t* __restrict__ out) {
  int idx = blockIdx.x * blockDim.x + threadIdx.x;
  if (idx >= BATCH * SEQ_LEN * D_MODEL) return;
  int e = idx % D_MODEL;
  int l = (idx / D_MODEL) % SEQ_LEN;
  int b = idx / (D_MODEL * SEQ_LEN);
  int h = e >> 6;
  int d = e & 63;
  out[(((size_t)(b * NUM_HEADS + h) * HEAD_DIM) + d) * SEQ_LEN + l] =
      (bf16_t)in[idx];
}

// ---------------------------------------------------------------- GEMM
// out[m,n] = sum_k A[m,k] * W[n,k]   (A: MxK bf16 row-major, W: NxK bf16)
// Block tile 128x128, K-step 32, 8 waves; wave tile 32x64 (2x4 WMMA frags).
// Double-buffered LDS fed by GLOBAL_LOAD_ASYNC_TO_LDS_B128 (ASYNCcnt).
constexpr int LDA_PAD = 40;   // padded LDS row (halves): 80B, 16B aligned

__global__ __launch_bounds__(256) void gemm_bf16_kernel(
    const bf16_t* __restrict__ A, const bf16_t* __restrict__ W,
    float* __restrict__ out, int M, int N, int K) {
  __shared__ bf16_t As[2][128 * LDA_PAD];
  __shared__ bf16_t Bs[2][128 * LDA_PAD];

  const int tid   = threadIdx.x;
  const int lane  = tid & 31;
  const int wave  = tid >> 5;
  const int wm    = wave & 3;          // 32-row chunk within block
  const int wn    = wave >> 2;         // 64-col chunk within block
  const int mBase = blockIdx.y * 128;
  const int nBase = blockIdx.x * 128;

  const int ldRow = tid >> 1;          // 0..127
  const int ldSeg = (tid & 1) * 16;    // halves: 0 or 16

  const int l15   = lane & 15;
  const int lhalf = lane >> 4;

  v8f acc[2][4];
#pragma unroll
  for (int i = 0; i < 2; ++i)
#pragma unroll
    for (int j = 0; j < 4; ++j) acc[i][j] = (v8f){0, 0, 0, 0, 0, 0, 0, 0};

  const bf16_t* gaRow = A + (size_t)(mBase + ldRow) * K + ldSeg;
  const bf16_t* gbRow = W + (size_t)(nBase + ldRow) * K + ldSeg;
  const unsigned saOff[2] = {
      (unsigned)(size_t)&As[0][ldRow * LDA_PAD + ldSeg],
      (unsigned)(size_t)&As[1][ldRow * LDA_PAD + ldSeg]};
  const unsigned sbOff[2] = {
      (unsigned)(size_t)&Bs[0][ldRow * LDA_PAD + ldSeg],
      (unsigned)(size_t)&Bs[1][ldRow * LDA_PAD + ldSeg]};

  // prologue: async-fill buffer 0 with the first K-slice
  async_copy_32B(saOff[0], gaRow);
  async_copy_32B(sbOff[0], gbRow);

  for (int k0 = 0; k0 < K; k0 += 32) {
    const int buf = (k0 >> 5) & 1;
    // my async copies for this tile have landed ...
    wait_async0();
    // ... and everyone has finished computing on the other buffer
    __syncthreads();

    if (k0 + 32 < K) {                 // overlap: fill next slice while we MAC
      async_copy_32B(saOff[buf ^ 1], gaRow + k0 + 32);
      async_copy_32B(sbOff[buf ^ 1], gbRow + k0 + 32);
      __builtin_prefetch(gaRow + k0 + 64, 0, 1);   // warm GL2 one slice ahead
      __builtin_prefetch(gbRow + k0 + 64, 0, 1);
    }

    // A fragments: interleaved K pattern -> two 16B chunks at lhalf*8, +16
    Frag16 af[2];
#pragma unroll
    for (int fm = 0; fm < 2; ++fm) {
      const bf16_t* src =
          &As[buf][(wm * 32 + fm * 16 + l15) * LDA_PAD + lhalf * 8];
      af[fm].u[0] = *reinterpret_cast<const uint4*>(src);
      af[fm].u[1] = *reinterpret_cast<const uint4*>(src + 16);
    }
    // B fragments: interleaved K pattern -> two 16B chunks at lhalf*8, +16
    Frag16 bf[4];
#pragma unroll
    for (int fn = 0; fn < 4; ++fn) {
      const bf16_t* src =
          &Bs[buf][(wn * 64 + fn * 16 + l15) * LDA_PAD + lhalf * 8];
      bf[fn].u[0] = *reinterpret_cast<const uint4*>(src);
      bf[fn].u[1] = *reinterpret_cast<const uint4*>(src + 16);
    }
#pragma unroll
    for (int fm = 0; fm < 2; ++fm)
#pragma unroll
      for (int fn = 0; fn < 4; ++fn)
        acc[fm][fn] = wmma_bf16(af[fm].v, bf[fn].v, acc[fm][fn]);
  }

#pragma unroll
  for (int fm = 0; fm < 2; ++fm)
#pragma unroll
    for (int fn = 0; fn < 4; ++fn) {
      int col = nBase + wn * 64 + fn * 16 + l15;
#pragma unroll
      for (int v = 0; v < 8; ++v) {
        int row = mBase + wm * 32 + fm * 16 + v + lhalf * 8;
        out[(size_t)row * N + col] = acc[fm][fn][v];
      }
    }
}

// ---------------------------------------------------------------- attention
// Flash-style causal attention.  grid = (B*H, L/64), block = 128 (4 waves).
// Each wave: 16 query rows; loops over 32-key blocks with online softmax.
__global__ __launch_bounds__(128) void attn_kernel(
    const bf16_t* __restrict__ Qb,   // (B,L,D) rope'd, pre-scaled by 1/8
    const bf16_t* __restrict__ Kb,   // (B,L,D) rope'd
    const bf16_t* __restrict__ Vt,   // (B,H,Dh,L) transposed
    bf16_t* __restrict__ Ob) {       // (B,L,D)
  __shared__ bf16_t Pbuf[4][16 * 32];   // per-wave P staging (1KB each)

  const int lane  = threadIdx.x & 31;
  const int wave  = threadIdx.x >> 5;
  const int b     = blockIdx.x >> 4;
  const int h     = blockIdx.x & 15;
  const int qrow0 = blockIdx.y * 64 + wave * 16;
  const int l15   = lane & 15;
  const int lhalf = lane >> 4;

  // Q A-fragments: row = qrow0 + l15, interleaved K chunks
  const bf16_t* qptr =
      Qb + ((size_t)(b * SEQ_LEN) + qrow0 + l15) * D_MODEL + h * HEAD_DIM;
  Frag16 q[2];
#pragma unroll
  for (int f = 0; f < 2; ++f) {
    q[f].u[0] = *reinterpret_cast<const uint4*>(qptr + f * 32 + lhalf * 8);
    q[f].u[1] = *reinterpret_cast<const uint4*>(qptr + f * 32 + lhalf * 8 + 16);
  }

  v8f o[4];
#pragma unroll
  for (int f = 0; f < 4; ++f) o[f] = (v8f){0, 0, 0, 0, 0, 0, 0, 0};
  float rmax[8], rsum[8];
#pragma unroll
  for (int v = 0; v < 8; ++v) { rmax[v] = -3.0e38f; rsum[v] = 0.0f; }

  const bf16_t* kbase = Kb + (size_t)(b * SEQ_LEN) * D_MODEL + h * HEAD_DIM;
  const bf16_t* vbase =
      Vt + ((size_t)(b * NUM_HEADS) + h) * HEAD_DIM * SEQ_LEN;
  bf16_t* pbuf = &Pbuf[wave][0];

  for (int kb = 0; kb <= qrow0 + 15; kb += 32) {
    // ---- S = Q * K^T  (two 16-key column tiles)
    v8f s[2];
#pragma unroll
    for (int t = 0; t < 2; ++t) {
      const bf16_t* kp = kbase + (size_t)(kb + t * 16 + l15) * D_MODEL;
      Frag16 kf0, kf1;                         // B-frags: interleaved K chunks
      kf0.u[0] = *reinterpret_cast<const uint4*>(kp + lhalf * 8);
      kf0.u[1] = *reinterpret_cast<const uint4*>(kp + lhalf * 8 + 16);
      kf1.u[0] = *reinterpret_cast<const uint4*>(kp + 32 + lhalf * 8);
      kf1.u[1] = *reinterpret_cast<const uint4*>(kp + 32 + lhalf * 8 + 16);
      v8f z = (v8f){0, 0, 0, 0, 0, 0, 0, 0};
      s[t] = wmma_bf16(q[0].v, kf0.v, z);
      s[t] = wmma_bf16(q[1].v, kf1.v, s[t]);
    }

    // ---- causal mask + online softmax (rows live in 16-lane halves)
#pragma unroll
    for (int v = 0; v < 8; ++v) {
      const int qidx = qrow0 + v + lhalf * 8;
#pragma unroll
      for (int t = 0; t < 2; ++t) {
        int kidx = kb + t * 16 + l15;
        if (kidx > qidx) s[t][v] = -3.0e38f;
      }
      float mx = fmaxf(s[0][v], s[1][v]);
#pragma unroll
      for (int m = 1; m < 16; m <<= 1) mx = fmaxf(mx, __shfl_xor(mx, m, 32));
      float mnew  = fmaxf(rmax[v], mx);
      float scale = __expf(rmax[v] - mnew);
      float p0 = __expf(s[0][v] - mnew);
      float p1 = __expf(s[1][v] - mnew);
      float ps = p0 + p1;
#pragma unroll
      for (int m = 1; m < 16; m <<= 1) ps += __shfl_xor(ps, m, 32);
      rsum[v] = rsum[v] * scale + ps;
      rmax[v] = mnew;
#pragma unroll
      for (int f = 0; f < 4; ++f) o[f][v] *= scale;
      // stage P (C-layout -> row-major LDS tile) as bf16
      int prow = v + lhalf * 8;
      pbuf[prow * 32 + l15]      = (bf16_t)p0;
      pbuf[prow * 32 + 16 + l15] = (bf16_t)p1;
    }
    asm volatile("s_wait_dscnt 0" ::: "memory");   // LDS RAW fence (same wave)

    // ---- reload P as A-fragment (16x32, interleaved K pattern)
    Frag16 pa;
    const bf16_t* pp = pbuf + l15 * 32 + lhalf * 8;
    pa.u[0] = *reinterpret_cast<const uint4*>(pp);
    pa.u[1] = *reinterpret_cast<const uint4*>(pp + 16);

    // ---- O += P * V   (V B-frags contiguous along keys thanks to transpose)
#pragma unroll
    for (int f = 0; f < 4; ++f) {
      const bf16_t* vp = vbase + (size_t)(f * 16 + l15) * SEQ_LEN + kb;
      Frag16 vf;
      vf.u[0] = *reinterpret_cast<const uint4*>(vp + lhalf * 8);
      vf.u[1] = *reinterpret_cast<const uint4*>(vp + lhalf * 8 + 16);
      o[f] = wmma_bf16(pa.v, vf.v, o[f]);
    }
  }

  // ---- normalize + store bf16 O in (B,L,D)
#pragma unroll
  for (int f = 0; f < 4; ++f)
#pragma unroll
    for (int v = 0; v < 8; ++v) {
      int row = qrow0 + v + lhalf * 8;
      float val = o[f][v] / rsum[v];
      Ob[((size_t)(b * SEQ_LEN) + row) * D_MODEL + h * HEAD_DIM + f * 16 + l15] =
          (bf16_t)val;
    }
}

// ---------------------------------------------------------------- launch
extern "C" void kernel_launch(void* const* d_in, const int* in_sizes, int n_in,
                              void* d_out, int out_size, void* d_ws,
                              size_t ws_size, hipStream_t stream) {
  const float* x  = (const float*)d_in[0];
  const float* Wq = (const float*)d_in[1];
  const float* Wk = (const float*)d_in[2];
  const float* Wv = (const float*)d_in[3];
  const float* Wo = (const float*)d_in[4];
  const int* pos  = (const int*)d_in[5];
  float* out      = (float*)d_out;

  const int NX = BATCH * SEQ_LEN * D_MODEL;  // 4M elems
  const int NW = D_MODEL * D_MODEL;          // 1M elems

  char* ws = (char*)d_ws;
  const size_t MB = 1024 * 1024;
  bf16_t* xb   = (bf16_t*)(ws + 0);          //  8 MB
  bf16_t* wqb  = (bf16_t*)(ws + 8 * MB);     //  2 MB
  bf16_t* wkb  = (bf16_t*)(ws + 10 * MB);    //  2 MB
  bf16_t* wvb  = (bf16_t*)(ws + 12 * MB);    //  2 MB
  bf16_t* wob  = (bf16_t*)(ws + 14 * MB);    //  2 MB
  float*  tmpf = (float*)(ws + 16 * MB);     // 16 MB fp32 GEMM output
  bf16_t* qb   = (bf16_t*)(ws + 32 * MB);    //  8 MB
  bf16_t* kbuf = (bf16_t*)(ws + 40 * MB);    //  8 MB
  bf16_t* vt   = (bf16_t*)(ws + 48 * MB);    //  8 MB
  bf16_t* ob   = (bf16_t*)(ws + 56 * MB);    //  8 MB  (total 64 MB)

  // 1) bf16 conversions
  cvt_f32_to_bf16_kernel<<<NX / 256, 256, 0, stream>>>(x, xb, NX);
  cvt_f32_to_bf16_kernel<<<NW / 256, 256, 0, stream>>>(Wq, wqb, NW);
  cvt_f32_to_bf16_kernel<<<NW / 256, 256, 0, stream>>>(Wk, wkb, NW);
  cvt_f32_to_bf16_kernel<<<NW / 256, 256, 0, stream>>>(Wv, wvb, NW);
  cvt_f32_to_bf16_kernel<<<NW / 256, 256, 0, stream>>>(Wo, wob, NW);

  dim3 ggrid(D_MODEL / 128, MROWS / 128);    // (8, 32)
  const int PAIRS = BATCH * SEQ_LEN * (D_MODEL / 2);

  // 2) Q projection + RoPE (fold 1/sqrt(64) = 0.125 into Q)
  gemm_bf16_kernel<<<ggrid, 256, 0, stream>>>(xb, wqb, tmpf, MROWS, D_MODEL,
                                              D_MODEL);
  rope_qk_kernel<<<PAIRS / 256, 256, 0, stream>>>(tmpf, qb, pos, 0.125f);

  // 3) K projection + RoPE
  gemm_bf16_kernel<<<ggrid, 256, 0, stream>>>(xb, wkb, tmpf, MROWS, D_MODEL,
                                              D_MODEL);
  rope_qk_kernel<<<PAIRS / 256, 256, 0, stream>>>(tmpf, kbuf, pos, 1.0f);

  // 4) V projection + transpose to (B,H,Dh,L)
  gemm_bf16_kernel<<<ggrid, 256, 0, stream>>>(xb, wvb, tmpf, MROWS, D_MODEL,
                                              D_MODEL);
  vtrans_kernel<<<NX / 256, 256, 0, stream>>>(tmpf, vt);

  // 5) flash attention
  dim3 agrid(BATCH * NUM_HEADS, SEQ_LEN / 64);   // (32, 32)
  attn_kernel<<<agrid, 128, 0, stream>>>(qb, kbuf, vt, ob);

  // 6) output projection -> fp32 d_out
  gemm_bf16_kernel<<<ggrid, 256, 0, stream>>>(ob, wob, out, MROWS, D_MODEL,
                                              D_MODEL);
}
